// GroupQueryAttention_87600152969633
// MI455X (gfx1250) — matrix operand order repaired and checked by reading, not compiled, not run on hardware
//
#include <hip/hip_runtime.h>
#include <hip/hip_bf16.h>

typedef __bf16 bf16_t;
typedef __bf16 v16bf __attribute__((ext_vector_type(16)));
typedef float  v8f   __attribute__((ext_vector_type(8)));

#define HIDDEN   2048
#define KVD      512
#define HEADS    32
#define GROUPS   8
#define HEAD_DIM 64
#define B_       2
#define S_       2048
#define M_TOTAL  (B_ * S_)

union ABFrag { v16bf v; unsigned u[8]; };
union U2H    { uint2 u; bf16_t h[4]; };

static __device__ inline v8f vzero() {
    v8f z;
#pragma unroll
    for (int i = 0; i < 8; ++i) z[i] = 0.0f;
    return z;
}

static __device__ inline v8f wmma_bf16(v16bf a, v16bf b, v8f c) {
    // D = A(16x32) * B(32x16) + C, f32 accumulate
    return __builtin_amdgcn_wmma_f32_16x16x32_bf16(
        /*neg_a=*/false, a, /*neg_b=*/false, b,
        /*c_mod=*/(short)0, c, /*reuse_a=*/false, /*reuse_b=*/false);
}

// Async global->LDS (CDNA5, ASYNCcnt-tracked). Per-lane 16B transfer.
// VDST carries the per-lane LDS byte address, VADDR the 64-bit global address.
static __device__ inline void async_ld_b128(void* lds_dst, const void* gsrc) {
    unsigned l = (unsigned)(uintptr_t)lds_dst;  // generic LDS ptr: low 32 bits = LDS addr
    asm volatile("global_load_async_to_lds_b128 %0, %1, off"
                 :: "v"(l), "v"(gsrc)
                 : "memory");
}
static __device__ inline void wait_async0() {
    asm volatile("s_wait_asynccnt 0" ::: "memory");
}

// 16-bit A matrix 16x32 fragment from LDS tile stored row-major [m][k], ld in halves.
static __device__ inline v16bf load_A_frag(const bf16_t* p, int ld, int row0, int k0, int lane) {
    const bf16_t* base = p + (size_t)(row0 + (lane & 15)) * ld + k0 + ((lane & 16) ? 8 : 0);
    ABFrag f;
#pragma unroll
    for (int v = 0; v < 4; ++v) {
        f.u[v]     = *(const unsigned*)(base + 2 * v);
        f.u[v + 4] = *(const unsigned*)(base + 16 + 2 * v);
    }
    return f.v;
}

// 16-bit B matrix 32x16 fragment from LDS tile stored TRANSPOSED [n][k], ld in halves.
static __device__ inline v16bf load_B_frag(const bf16_t* p, int ld, int n0, int k0, int lane) {
    const bf16_t* base = p + (size_t)(n0 + (lane & 15)) * ld + k0 + ((lane & 16) ? 8 : 0);
    ABFrag f;
#pragma unroll
    for (int v = 0; v < 4; ++v) {
        f.u[v]     = *(const unsigned*)(base + 2 * v);
        f.u[v + 4] = *(const unsigned*)(base + 16 + 2 * v);
    }
    return f.v;
}

// ---------------------------------------------------------------------------
// Tiled GEMM + bias:  C[M,N] = A[M,K] @ W[K,N] + bias[N]
// BM=BN=128, BK=32, 256 threads (8 waves, 4x2 wave grid, wave tile 32x64).
// ---------------------------------------------------------------------------
template <bool A_BF16, bool OUT_BF16>
__global__ __launch_bounds__(256) void gemm_bias_wmma(
    const void* __restrict__ Av, const float* __restrict__ W,
    const float* __restrict__ bias, void* __restrict__ Cv,
    int M, int N, int K) {
    constexpr int BM = 128, BN = 128, BK = 32, LDT = BK + 8;
    __shared__ __align__(16) bf16_t As[BM * LDT];
    __shared__ __align__(16) bf16_t Bs[BN * LDT];

    const int tid  = threadIdx.x;
    const int lane = tid & 31;
    const int wave = tid >> 5;
    const int wm   = (wave & 3) * 32;  // wave M offset
    const int wn   = (wave >> 2) * 64; // wave N offset
    const int m0   = blockIdx.y * BM;
    const int n0   = blockIdx.x * BN;

    v8f acc[2][4];
#pragma unroll
    for (int mi = 0; mi < 2; ++mi)
#pragma unroll
        for (int ni = 0; ni < 4; ++ni) acc[mi][ni] = vzero();

    const int rowA = tid >> 3;        // 0..31
    const int colA = (tid & 7) * 4;   // 0..28, 4 elems
    const int kRow = tid >> 5;        // 0..7
    const int nQ   = (tid & 31) * 4;  // 0..124, 4 cols

    for (int kk = 0; kk < K; kk += BK) {
        // ---- stage A tile [128 x 32] ----
        if (A_BF16) {
            const bf16_t* A = (const bf16_t*)Av;
#pragma unroll
            for (int p = 0; p < 4; ++p) {
                int row = p * 32 + rowA;
                uint2 d = *(const uint2*)(A + (size_t)(m0 + row) * K + kk + colA);
                *(uint2*)&As[row * LDT + colA] = d;
            }
        } else {
            const float* A = (const float*)Av;
#pragma unroll
            for (int p = 0; p < 4; ++p) {
                int row = p * 32 + rowA;
                float4 d = *(const float4*)(A + (size_t)(m0 + row) * K + kk + colA);
                bf16_t* dst = &As[row * LDT + colA];
                dst[0] = (bf16_t)d.x; dst[1] = (bf16_t)d.y;
                dst[2] = (bf16_t)d.z; dst[3] = (bf16_t)d.w;
            }
        }
        // ---- stage W tile [32 x 128] transposed -> Bs[n][k] ----
#pragma unroll
        for (int p = 0; p < 4; ++p) {
            int k = p * 8 + kRow;
            float4 d = *(const float4*)(W + (size_t)(kk + k) * N + n0 + nQ);
            Bs[(nQ + 0) * LDT + k] = (bf16_t)d.x;
            Bs[(nQ + 1) * LDT + k] = (bf16_t)d.y;
            Bs[(nQ + 2) * LDT + k] = (bf16_t)d.z;
            Bs[(nQ + 3) * LDT + k] = (bf16_t)d.w;
        }
        if (kk + BK < K) {
            __builtin_prefetch((const char*)(W + (size_t)(kk + BK + kRow) * N + n0 + nQ), 0, 1);
        }
        __syncthreads();

        v16bf a0 = load_A_frag(As, LDT, wm, 0, lane);
        v16bf a1 = load_A_frag(As, LDT, wm + 16, 0, lane);
#pragma unroll
        for (int ni = 0; ni < 4; ++ni) {
            v16bf bfr = load_B_frag(Bs, LDT, wn + ni * 16, 0, lane);
            acc[0][ni] = wmma_bf16(a0, bfr, acc[0][ni]);
            acc[1][ni] = wmma_bf16(a1, bfr, acc[1][ni]);
        }
        __syncthreads();
    }

    // ---- epilogue: bias + store ----
#pragma unroll
    for (int mi = 0; mi < 2; ++mi) {
        int rowb = m0 + wm + mi * 16 + ((lane & 16) ? 8 : 0);
#pragma unroll
        for (int ni = 0; ni < 4; ++ni) {
            int col  = n0 + wn + ni * 16 + (lane & 15);
            float bv = bias[col];
#pragma unroll
            for (int r = 0; r < 8; ++r) {
                float v   = acc[mi][ni][r] + bv;
                size_t ix = (size_t)(rowb + r) * N + col;
                if (OUT_BF16) ((bf16_t*)Cv)[ix] = (bf16_t)v;
                else          ((float*)Cv)[ix]  = v;
            }
        }
    }
}

// ---------------------------------------------------------------------------
// Flash attention over bf16 q/k/v (grouped query). One block per
// (batch*head, 64-query tile); 4 waves, each owning 16 query rows.
// K tile staged via CDNA5 async global->LDS; V staged transposed manually.
// ---------------------------------------------------------------------------
__global__ __launch_bounds__(128) void gqa_flash_attn(
    const bf16_t* __restrict__ qb, const bf16_t* __restrict__ kb,
    const bf16_t* __restrict__ vb, bf16_t* __restrict__ attn) {
    constexpr int LDK = 72;                               // 144 B row stride (16B-mult)
    __shared__ __align__(16) bf16_t Kt[64 * LDK];         // [kv][d]
    __shared__ __align__(16) bf16_t Vt[64 * LDK];         // [d][kv] (transposed)
    __shared__ __align__(16) bf16_t Pb[4 * 16 * LDK];     // per-wave P staging

    const int tid  = threadIdx.x;
    const int lane = tid & 31;
    const int wave = tid >> 5;
    const int bh   = blockIdx.x;
    const int b    = bh >> 5;
    const int h    = bh & 31;
    const int g    = h >> 2;                // kv group (REP = 4)
    const int hOff  = h * HEAD_DIM;         // into HIDDEN
    const int kvOff = g * HEAD_DIM;         // into KVD
    const int q0    = blockIdx.y * 64 + wave * 16;
    const float SCALE = 0.125f;             // 1/sqrt(64)

    // Q fragments (A layout) directly from global bf16 q
    const bf16_t* qrow = qb + ((size_t)(b * S_) + q0 + (lane & 15)) * HIDDEN + hOff;
    const int kbse = (lane & 16) ? 8 : 0;
    ABFrag aq0, aq1;
#pragma unroll
    for (int v = 0; v < 4; ++v) {
        aq0.u[v]     = *(const unsigned*)(qrow + kbse + 2 * v);
        aq0.u[v + 4] = *(const unsigned*)(qrow + kbse + 16 + 2 * v);
        aq1.u[v]     = *(const unsigned*)(qrow + 32 + kbse + 2 * v);
        aq1.u[v + 4] = *(const unsigned*)(qrow + 32 + kbse + 16 + 2 * v);
    }

    float mrun[8], lrun[8];
    v8f acc[4];
#pragma unroll
    for (int r = 0; r < 8; ++r) { mrun[r] = -3.0e38f; lrun[r] = 0.0f; }
#pragma unroll
    for (int j = 0; j < 4; ++j) acc[j] = vzero();

    const int krow8 = tid >> 3;       // K async loader: 0..15 row-in-pass
    const int kcol8 = (tid & 7) * 8;  // 8 halves = 16 B per lane
    const int vkv   = tid >> 1;       // V loader: kv row 0..63
    const int vhi   = (tid & 1) * 32; // half of d range

    for (int kt = 0; kt < S_; kt += 64) {
        // ---- stage K tile [kv][d] via async global->LDS (ASYNCcnt) ----
#pragma unroll
        for (int p = 0; p < 4; ++p) {
            int kv = p * 16 + krow8;
            async_ld_b128(&Kt[kv * LDK + kcol8],
                          kb + ((size_t)(b * S_) + kt + kv) * KVD + kvOff + kcol8);
        }
        // ---- stage V tile transposed [d][kv] ----
#pragma unroll
        for (int j = 0; j < 8; ++j) {
            int d0 = vhi + j * 4;
            U2H u;
            u.u = *(const uint2*)(vb + ((size_t)(b * S_) + kt + vkv) * KVD + kvOff + d0);
#pragma unroll
            for (int q = 0; q < 4; ++q) Vt[(d0 + q) * LDK + vkv] = u.h[q];
        }
        wait_async0();
        __syncthreads();

        // ---- scores: S = Q @ K^T  (per wave: 16 x 64) ----
        v8f sc[4];
#pragma unroll
        for (int j = 0; j < 4; ++j) {
            sc[j] = vzero();
            v16bf b0 = load_B_frag(Kt, LDK, j * 16, 0, lane);
            v16bf b1 = load_B_frag(Kt, LDK, j * 16, 32, lane);
            sc[j] = wmma_bf16(aq0.v, b0, sc[j]);
            sc[j] = wmma_bf16(aq1.v, b1, sc[j]);
        }

        // ---- online softmax (rows live in element r, 16-lane half-groups) ----
        float mloc[8];
#pragma unroll
        for (int r = 0; r < 8; ++r) {
            sc[0][r] *= SCALE; sc[1][r] *= SCALE; sc[2][r] *= SCALE; sc[3][r] *= SCALE;
            mloc[r] = fmaxf(fmaxf(sc[0][r], sc[1][r]), fmaxf(sc[2][r], sc[3][r]));
        }
#pragma unroll
        for (int off = 1; off < 16; off <<= 1)
#pragma unroll
            for (int r = 0; r < 8; ++r)
                mloc[r] = fmaxf(mloc[r], __shfl_xor(mloc[r], off, 32));

        float corr[8], rsum[8];
#pragma unroll
        for (int r = 0; r < 8; ++r) {
            float mnew = fmaxf(mrun[r], mloc[r]);
            corr[r] = __expf(mrun[r] - mnew);
            mrun[r] = mnew;
        }
#pragma unroll
        for (int j = 0; j < 4; ++j)
#pragma unroll
            for (int r = 0; r < 8; ++r)
                sc[j][r] = __expf(sc[j][r] - mrun[r]);
#pragma unroll
        for (int r = 0; r < 8; ++r)
            rsum[r] = sc[0][r] + sc[1][r] + sc[2][r] + sc[3][r];
#pragma unroll
        for (int off = 1; off < 16; off <<= 1)
#pragma unroll
            for (int r = 0; r < 8; ++r)
                rsum[r] += __shfl_xor(rsum[r], off, 32);
#pragma unroll
        for (int r = 0; r < 8; ++r) lrun[r] = lrun[r] * corr[r] + rsum[r];
#pragma unroll
        for (int dj = 0; dj < 4; ++dj)
#pragma unroll
            for (int r = 0; r < 8; ++r) acc[dj][r] *= corr[r];

        // ---- P: C-layout -> LDS -> A-layout (per-wave, LDS is in-order) ----
        {
            int prow = wave * 16 + ((lane & 16) ? 8 : 0);
            int pcol = lane & 15;
#pragma unroll
            for (int j = 0; j < 4; ++j)
#pragma unroll
                for (int r = 0; r < 8; ++r)
                    Pb[(size_t)(prow + r) * LDK + j * 16 + pcol] = (bf16_t)sc[j][r];
        }
        v16bf pa0 = load_A_frag(&Pb[(size_t)(wave * 16) * LDK], LDK, 0, 0, lane);
        v16bf pa1 = load_A_frag(&Pb[(size_t)(wave * 16) * LDK], LDK, 0, 32, lane);

        // ---- O += P @ V ----
#pragma unroll
        for (int dj = 0; dj < 4; ++dj) {
            v16bf b0 = load_B_frag(Vt, LDK, dj * 16, 0, lane);
            v16bf b1 = load_B_frag(Vt, LDK, dj * 16, 32, lane);
            acc[dj] = wmma_bf16(pa0, b0, acc[dj]);
            acc[dj] = wmma_bf16(pa1, b1, acc[dj]);
        }
        __syncthreads();
    }

    // ---- normalize and store bf16 into [b, s, HIDDEN] at head offset ----
#pragma unroll
    for (int dj = 0; dj < 4; ++dj) {
#pragma unroll
        for (int r = 0; r < 8; ++r) {
            int row = q0 + r + ((lane & 16) ? 8 : 0);
            int d   = dj * 16 + (lane & 15);
            attn[((size_t)(b * S_) + row) * HIDDEN + hOff + d] =
                (bf16_t)(acc[dj][r] / lrun[r]);
        }
    }
}

// ---------------------------------------------------------------------------
extern "C" void kernel_launch(void* const* d_in, const int* in_sizes, int n_in,
                              void* d_out, int out_size, void* d_ws, size_t ws_size,
                              hipStream_t stream) {
    (void)in_sizes; (void)n_in; (void)out_size; (void)ws_size;
    const float* x  = (const float*)d_in[0];
    const float* eo = (const float*)d_in[1];
    const float* Wq = (const float*)d_in[2];
    const float* bq = (const float*)d_in[3];
    const float* Wk = (const float*)d_in[4];
    const float* bk = (const float*)d_in[5];
    const float* Wv = (const float*)d_in[6];
    const float* bv = (const float*)d_in[7];
    const float* Wo = (const float*)d_in[8];
    const float* bo = (const float*)d_in[9];
    float* out = (float*)d_out;

    // workspace: q(16MB) | k(4MB) | v(4MB) | attn(16MB), all bf16
    bf16_t* qb   = (bf16_t*)d_ws;
    bf16_t* kb   = qb + (size_t)M_TOTAL * HIDDEN;
    bf16_t* vb   = kb + (size_t)M_TOTAL * KVD;
    bf16_t* attn = vb + (size_t)M_TOTAL * KVD;

    dim3 blk(256);
    gemm_bias_wmma<false, true><<<dim3(HIDDEN / 128, M_TOTAL / 128), blk, 0, stream>>>(
        x, Wq, bq, qb, M_TOTAL, HIDDEN, HIDDEN);
    gemm_bias_wmma<false, true><<<dim3(KVD / 128, M_TOTAL / 128), blk, 0, stream>>>(
        eo, Wk, bk, kb, M_TOTAL, KVD, HIDDEN);
    gemm_bias_wmma<false, true><<<dim3(KVD / 128, M_TOTAL / 128), blk, 0, stream>>>(
        eo, Wv, bv, vb, M_TOTAL, KVD, HIDDEN);
    gqa_flash_attn<<<dim3(B_ * HEADS, S_ / 64), dim3(128), 0, stream>>>(qb, kb, vb, attn);
    gemm_bias_wmma<true, false><<<dim3(HIDDEN / 128, M_TOTAL / 128), blk, 0, stream>>>(
        attn, Wo, bo, out, M_TOTAL, HIDDEN, HIDDEN);
}
